// Attention_lorad_33002528702628
// MI455X (gfx1250) — matrix operand order repaired and checked by reading, not compiled, not run on hardware
//
#include <hip/hip_runtime.h>
#include <math.h>

// ---------------------------------------------------------------------------
// Types for CDNA5 WMMA (wave32): 16x16x32 f16 -> f32
// ---------------------------------------------------------------------------
typedef __attribute__((ext_vector_type(16))) _Float16 v16h;
typedef __attribute__((ext_vector_type(8)))  _Float16 v8h;
typedef __attribute__((ext_vector_type(8)))  float    v8f;

union V16U { v16h v; v8h h[2]; };

// A-matrix 16x32 f16 layout (ISA 7.12.2): lane L(0..15)=row M=L holds K = hi*8+0..7
// and K = 16+hi*8+0..7 (hi = lane>>4). Two contiguous 16B loads.
__device__ __forceinline__ v16h load_a_frag(const _Float16* __restrict__ row, int k, int hi) {
  V16U u;
  u.h[0] = *(const v8h*)(row + k + hi * 8);
  u.h[1] = *(const v8h*)(row + k + 16 + hi * 8);
  return u.v;
}
__device__ __forceinline__ v16h load_a_frag_lds(const _Float16* row, int k, int hi) {
  V16U u;
  u.h[0] = *(const v8h*)(row + k + hi * 8);
  u.h[1] = *(const v8h*)(row + k + 16 + hi * 8);
  return u.v;
}
// B-matrix 32x16 f16 layout: lane L(0..15)=col N=L holds K = hi*8+0..7 and K = 16+hi*8+0..7
// (same K order as A) when B is stored [N][K] (i.e. y = x*W^T reads W rows directly).
__device__ __forceinline__ v16h load_b_frag(const _Float16* __restrict__ row, int k, int hi) {
  V16U u;
  u.h[0] = *(const v8h*)(row + k + hi * 8);
  u.h[1] = *(const v8h*)(row + k + 16 + hi * 8);
  return u.v;
}
__device__ __forceinline__ v8f wmma32(v16h a, v16h b, v8f c) {
  return __builtin_amdgcn_wmma_f32_16x16x32_f16(false, a, false, b, (short)0, c, false, false);
}

#define DIMC      768
#define NROWS     8192          // 8 * 1024
#define THREE_C   2304
#define HEADS     12
#define HDIM      64
#define SEQ       1024
#define BH        96            // 8 * 12
#define LORA_SC   0.0625f       // 1/R
#define SOFT_SC   0.125f        // 64^-0.5

// ---------------------------------------------------------------------------
// Staging kernels
// ---------------------------------------------------------------------------
__global__ void prep_x_kernel(const float* __restrict__ x, _Float16* __restrict__ X16, int n) {
  int idx = blockIdx.x * blockDim.x + threadIdx.x;
  if (idx < n) X16[idx] = (_Float16)x[idx];
}

__global__ void prep_w_kernel(const float* __restrict__ qkv_w, const float* __restrict__ qkv_b,
                              const float* __restrict__ q_w,   const float* __restrict__ q_b,
                              const float* __restrict__ v_w,   const float* __restrict__ v_b,
                              const float* __restrict__ q_A,   const float* __restrict__ v_A,
                              const float* __restrict__ q_B,   const float* __restrict__ v_B,
                              const float* __restrict__ proj_w,
                              _Float16* __restrict__ W16, _Float16* __restrict__ PW16,
                              float* __restrict__ BIAS, _Float16* __restrict__ BL16,
                              _Float16* __restrict__ A16) {
  int idx = blockIdx.x * blockDim.x + threadIdx.x;
  if (idx >= THREE_C * DIMC) return;
  int i = idx / DIMC, k = idx % DIMC;
  float wv;
  if (i < 768)       wv = qkv_w[(768 + i) * DIMC + k];   // K projection slice C:2C
  else if (i < 1536) wv = q_w[(i - 768) * DIMC + k];
  else               wv = v_w[(i - 1536) * DIMC + k];
  W16[idx] = (_Float16)wv;
  if (idx < DIMC * DIMC) PW16[idx] = (_Float16)proj_w[idx];
  if (idx < 32 * DIMC) {                                  // LoRA A stacked [q_A; v_A]
    A16[idx] = (_Float16)((i < 16) ? q_A[idx] : v_A[idx - 16 * DIMC]);
  }
  if (idx < THREE_C) {
    BIAS[idx] = (idx < 768) ? qkv_b[768 + idx] : (idx < 1536) ? q_b[idx - 768] : v_b[idx - 1536];
  }
  if (idx < 1536 * 32) {                                  // LoRA B (pre-scaled, padded K16->32)
    int n = idx >> 5, kk = idx & 31;
    float bv = 0.f;
    if (kk < 16) bv = ((n < 768) ? q_B[n * 16 + kk] : v_B[(n - 768) * 16 + kk]) * LORA_SC;
    BL16[idx] = (_Float16)bv;
  }
}

// ---------------------------------------------------------------------------
// LoRA A-product as skinny WMMA GEMM: XAraw = X16 @ A16^T  ([8192 x 32]).
// One wave per 64x32 tile (128 waves). Epilogue emits the K=32-padded XA16
// layout: cols [0..15]=x@qA^T, [16..31]=0, [32..47]=x@vA^T, [48..63]=0.
// ---------------------------------------------------------------------------
__global__ __launch_bounds__(256) void lora_a_gemm_kernel(
    const _Float16* __restrict__ X16, const _Float16* __restrict__ A16,
    _Float16* __restrict__ XA16) {
  const int lane = threadIdx.x & 31;
  const int l = lane & 15, hi = lane >> 4;
  const int wid = (blockIdx.x * 256 + threadIdx.x) >> 5;  // 0..127
  const int mbase = wid * 64;

  v8f acc[4][2] = {};
  for (int k = 0; k < DIMC; k += 32) {
    v16h af[4];
#pragma unroll
    for (int i = 0; i < 4; ++i)
      af[i] = load_a_frag(X16 + (size_t)(mbase + 16 * i + l) * DIMC, k, hi);
#pragma unroll
    for (int j = 0; j < 2; ++j) {
      v16h bf = load_b_frag(A16 + (size_t)(16 * j + l) * DIMC, k, hi);
#pragma unroll
      for (int i = 0; i < 4; ++i) acc[i][j] = wmma32(af[i], bf, acc[i][j]);
    }
  }
#pragma unroll
  for (int i = 0; i < 4; ++i)
#pragma unroll
    for (int r = 0; r < 8; ++r) {
      const size_t m = (size_t)(mbase + 16 * i + r + 8 * hi) * 64;
      XA16[m + l]      = (_Float16)acc[i][0][r];  // q part -> cols 0..15
      XA16[m + 16 + l] = (_Float16)0.f;           // pad
      XA16[m + 32 + l] = (_Float16)acc[i][1][r];  // v part -> cols 32..47
      XA16[m + 48 + l] = (_Float16)0.f;           // pad
    }
}

// ---------------------------------------------------------------------------
// Fused K/Q/V projection GEMM: [8192 x 768] @ W16^T -> [8192 x 2304]
// Wave = 64x64 tile. LoRA folded as one extra K=32 WMMA step for Q/V regions.
// Epilogue scatters f16 into Q16/K16 [bh][n][d] and VT16 [bh][d][n].
// ---------------------------------------------------------------------------
__global__ __launch_bounds__(256) void gemm_qkv_kernel(
    const _Float16* __restrict__ X16, const _Float16* __restrict__ W16,
    const float* __restrict__ BIAS, const _Float16* __restrict__ XA16,
    const _Float16* __restrict__ BL16,
    _Float16* __restrict__ K16, _Float16* __restrict__ Q16, _Float16* __restrict__ VT16) {
  const int lane = threadIdx.x & 31;
  const int l = lane & 15, hi = lane >> 4;
  const int wid = (blockIdx.x * 256 + threadIdx.x) >> 5;
  const int NT = THREE_C / 64;                 // 36
  const int mt = wid / NT, nt = wid % NT;
  const int mbase = mt * 64, nbase = nt * 64;

  v8f acc[4][4] = {};
  for (int k = 0; k < DIMC; k += 32) {
    v16h af[4];
#pragma unroll
    for (int i = 0; i < 4; ++i)
      af[i] = load_a_frag(X16 + (size_t)(mbase + 16 * i + l) * DIMC, k, hi);
#pragma unroll
    for (int j = 0; j < 4; ++j) {
      v16h bf = load_b_frag(W16 + (size_t)(nbase + 16 * j + l) * DIMC, k, hi);
#pragma unroll
      for (int i = 0; i < 4; ++i) acc[i][j] = wmma32(af[i], bf, acc[i][j]);
    }
  }

  // wave-uniform region select, forced scalar (0=K, 1=Q, 2=V)
  const int region = __builtin_amdgcn_readfirstlane(nbase / 768);
  if (region > 0) {                            // LoRA: + (xA^T)(B^T/R) as one K=32 step
    const int xaoff = (region == 1) ? 0 : 32;
    const int blbase = (region == 1) ? (nbase - 768) : (nbase - 1536 + 768);
    v16h af[4];
#pragma unroll
    for (int i = 0; i < 4; ++i)
      af[i] = load_a_frag(XA16 + (size_t)(mbase + 16 * i + l) * 64 + xaoff, 0, hi);
#pragma unroll
    for (int j = 0; j < 4; ++j) {
      v16h bf = load_b_frag(BL16 + (size_t)(blbase + 16 * j + l) * 32, 0, hi);
#pragma unroll
      for (int i = 0; i < 4; ++i) acc[i][j] = wmma32(af[i], bf, acc[i][j]);
    }
  }

#pragma unroll
  for (int i = 0; i < 4; ++i)
#pragma unroll
    for (int j = 0; j < 4; ++j) {
      const int n = nbase + 16 * j + l;
      const float bias = BIAS[n];
#pragma unroll
      for (int r = 0; r < 8; ++r) {
        const int m = mbase + 16 * i + r + 8 * hi;
        const float val = acc[i][j][r] + bias;
        const int b = m >> 10, mloc = m & 1023;
        if (region == 0) {
          const int h = n >> 6, d = n & 63;
          K16[(size_t)((b * HEADS + h) * SEQ + mloc) * HDIM + d] = (_Float16)val;
        } else if (region == 1) {
          const int nn = n - 768, h = nn >> 6, d = nn & 63;
          Q16[(size_t)((b * HEADS + h) * SEQ + mloc) * HDIM + d] = (_Float16)val;
        } else {
          const int nn = n - 1536, h = nn >> 6, d = nn & 63;
          VT16[(size_t)((b * HEADS + h) * HDIM + d) * SEQ + mloc] = (_Float16)val;
        }
      }
    }
}

// ---------------------------------------------------------------------------
// Flash attention: one wave per (bh, 16 query rows). Online softmax via
// __shfl_xor butterflies; P re-laid out D->A through per-wave LDS.
// ---------------------------------------------------------------------------
__global__ __launch_bounds__(256) void attn_kernel(
    const _Float16* __restrict__ Q16, const _Float16* __restrict__ K16,
    const _Float16* __restrict__ VT16, _Float16* __restrict__ O16) {
  __shared__ __align__(16) _Float16 lds_p[8][16 * 64];
  const int lane = threadIdx.x & 31;
  const int l = lane & 15, hi = lane >> 4;
  const int w = threadIdx.x >> 5;
  const int wid = blockIdx.x * 8 + w;
  const int bh = wid >> 6;                      // 96 batch-heads
  const int mt = wid & 63;                      // 64 query tiles of 16

  const _Float16* qrow = Q16 + (size_t)(bh * SEQ + mt * 16 + l) * HDIM;
  v16h qf0 = load_a_frag(qrow, 0, hi);
  v16h qf1 = load_a_frag(qrow, 32, hi);

  float mrun[8], lrun[8];
  v8f o[4] = {};
#pragma unroll
  for (int r = 0; r < 8; ++r) { mrun[r] = -1e30f; lrun[r] = 0.f; }

  for (int kb = 0; kb < SEQ; kb += 64) {
    v8f s[4] = {};
#pragma unroll
    for (int j = 0; j < 4; ++j) {
      const _Float16* krow = K16 + (size_t)(bh * SEQ + kb + 16 * j + l) * HDIM;
      s[j] = wmma32(qf0, load_b_frag(krow, 0, hi), s[j]);
      s[j] = wmma32(qf1, load_b_frag(krow, 32, hi), s[j]);
    }
    // online softmax: row r lives in lanes 0-15 (rows 0-7) / 16-31 (rows 8-15)
#pragma unroll
    for (int r = 0; r < 8; ++r) {
      float sv[4];
#pragma unroll
      for (int j = 0; j < 4; ++j) sv[j] = s[j][r] * SOFT_SC;
      float mx = fmaxf(fmaxf(sv[0], sv[1]), fmaxf(sv[2], sv[3]));
#pragma unroll
      for (int msk = 1; msk < 16; msk <<= 1) mx = fmaxf(mx, __shfl_xor(mx, msk, 32));
      const float mnew = fmaxf(mrun[r], mx);
      const float alpha = __expf(mrun[r] - mnew);
      mrun[r] = mnew;
      float rs = 0.f;
      const int base = (r + 8 * hi) * 64 + l;
#pragma unroll
      for (int j = 0; j < 4; ++j) {
        const float p = __expf(sv[j] - mnew);
        rs += p;
        lds_p[w][base + 16 * j] = (_Float16)p;
      }
#pragma unroll
      for (int msk = 1; msk < 16; msk <<= 1) rs += __shfl_xor(rs, msk, 32);
      lrun[r] = lrun[r] * alpha + rs;
#pragma unroll
      for (int j = 0; j < 4; ++j) o[j][r] *= alpha;
    }
    asm volatile("s_wait_dscnt 0" ::: "memory");   // LDS store -> load within wave
    v16h pf0 = load_a_frag_lds(&lds_p[w][l * 64], 0, hi);
    v16h pf1 = load_a_frag_lds(&lds_p[w][l * 64], 32, hi);
#pragma unroll
    for (int j = 0; j < 4; ++j) {
      const _Float16* vrow = VT16 + (size_t)(bh * HDIM + 16 * j + l) * SEQ + kb;
      o[j] = wmma32(pf0, load_b_frag(vrow, 0, hi), o[j]);
      o[j] = wmma32(pf1, load_b_frag(vrow, 32, hi), o[j]);
    }
  }

  const int b = bh / HEADS, h = bh % HEADS;
#pragma unroll
  for (int r = 0; r < 8; ++r) {
    const float inv = 1.0f / lrun[r];
    const int m = mt * 16 + r + 8 * hi;
    const size_t rowoff = (size_t)(b * SEQ + m) * DIMC + h * HDIM;
#pragma unroll
    for (int j = 0; j < 4; ++j) O16[rowoff + 16 * j + l] = (_Float16)(o[j][r] * inv);
  }
}

// ---------------------------------------------------------------------------
// Output projection: out = O16 @ proj_w^T + proj_b  (f32 store)
// ---------------------------------------------------------------------------
__global__ __launch_bounds__(256) void gemm_proj_kernel(
    const _Float16* __restrict__ O16, const _Float16* __restrict__ PW16,
    const float* __restrict__ pb, float* __restrict__ out) {
  const int lane = threadIdx.x & 31;
  const int l = lane & 15, hi = lane >> 4;
  const int wid = (blockIdx.x * 256 + threadIdx.x) >> 5;
  const int NT = DIMC / 64;                    // 12
  const int mt = wid / NT, nt = wid % NT;
  const int mbase = mt * 64, nbase = nt * 64;

  v8f acc[4][4] = {};
  for (int k = 0; k < DIMC; k += 32) {
    v16h af[4];
#pragma unroll
    for (int i = 0; i < 4; ++i)
      af[i] = load_a_frag(O16 + (size_t)(mbase + 16 * i + l) * DIMC, k, hi);
#pragma unroll
    for (int j = 0; j < 4; ++j) {
      v16h bf = load_b_frag(PW16 + (size_t)(nbase + 16 * j + l) * DIMC, k, hi);
#pragma unroll
      for (int i = 0; i < 4; ++i) acc[i][j] = wmma32(af[i], bf, acc[i][j]);
    }
  }
#pragma unroll
  for (int i = 0; i < 4; ++i)
#pragma unroll
    for (int j = 0; j < 4; ++j) {
      const int n = nbase + 16 * j + l;
      const float bias = pb[n];
#pragma unroll
      for (int r = 0; r < 8; ++r) {
        const int m = mbase + 16 * i + r + 8 * hi;
        out[(size_t)m * DIMC + n] = acc[i][j][r] + bias;
      }
    }
}

// ---------------------------------------------------------------------------
// Launch
// ---------------------------------------------------------------------------
extern "C" void kernel_launch(void* const* d_in, const int* in_sizes, int n_in,
                              void* d_out, int out_size, void* d_ws, size_t ws_size,
                              hipStream_t stream) {
  const float* x      = (const float*)d_in[0];
  const float* qkv_w  = (const float*)d_in[1];
  const float* qkv_b  = (const float*)d_in[2];
  const float* q_w    = (const float*)d_in[3];
  const float* q_b    = (const float*)d_in[4];
  const float* q_A    = (const float*)d_in[5];
  const float* q_B    = (const float*)d_in[6];
  const float* v_w    = (const float*)d_in[7];
  const float* v_b    = (const float*)d_in[8];
  const float* v_A    = (const float*)d_in[9];
  const float* v_B    = (const float*)d_in[10];
  const float* proj_w = (const float*)d_in[11];
  const float* proj_b = (const float*)d_in[12];
  float* out = (float*)d_out;

  // workspace layout (all chunks 256B multiples)
  char* ws = (char*)d_ws;
  size_t off = 0;
  auto take = [&](size_t bytes) { char* p = ws + off; off += bytes; return p; };
  _Float16* X16  = (_Float16*)take((size_t)NROWS * DIMC * 2);       // 12.6 MB
  _Float16* W16  = (_Float16*)take((size_t)THREE_C * DIMC * 2);     // 3.5 MB
  _Float16* PW16 = (_Float16*)take((size_t)DIMC * DIMC * 2);        // 1.2 MB
  float*    BIAS = (float*)   take((size_t)THREE_C * 4);
  _Float16* XA16 = (_Float16*)take((size_t)NROWS * 64 * 2);         // 1.0 MB
  _Float16* BL16 = (_Float16*)take((size_t)1536 * 32 * 2);
  _Float16* A16  = (_Float16*)take((size_t)32 * DIMC * 2);
  _Float16* Q16  = (_Float16*)take((size_t)BH * SEQ * HDIM * 2);    // 12.6 MB
  _Float16* K16  = (_Float16*)take((size_t)BH * SEQ * HDIM * 2);    // 12.6 MB
  _Float16* VT16 = (_Float16*)take((size_t)BH * HDIM * SEQ * 2);    // 12.6 MB
  _Float16* O16  = (_Float16*)take((size_t)NROWS * DIMC * 2);       // 12.6 MB

  prep_x_kernel<<<(NROWS * DIMC) / 256, 256, 0, stream>>>(x, X16, NROWS * DIMC);
  prep_w_kernel<<<(THREE_C * DIMC) / 256, 256, 0, stream>>>(
      qkv_w, qkv_b, q_w, q_b, v_w, v_b, q_A, v_A, q_B, v_B, proj_w,
      W16, PW16, BIAS, BL16, A16);

  // LoRA A-product: 128 waves -> 16 blocks of 8 waves
  lora_a_gemm_kernel<<<16, 256, 0, stream>>>(X16, A16, XA16);
  // 128 m-tiles * 36 n-tiles = 4608 waves -> 576 blocks of 8 waves
  gemm_qkv_kernel<<<576, 256, 0, stream>>>(X16, W16, BIAS, XA16, BL16, K16, Q16, VT16);
  // 96 bh * 64 query tiles = 6144 waves -> 768 blocks
  attn_kernel<<<768, 256, 0, stream>>>(Q16, K16, VT16, O16);
  // 128 * 12 = 1536 waves -> 192 blocks
  gemm_proj_kernel<<<192, 256, 0, stream>>>(O16, PW16, proj_b, out);
}
